// InfluenceProp_72121090835062
// MI455X (gfx1250) — compile-verified, matrix operand order checked by reading, not run on hardware
//
#include <hip/hip_runtime.h>
#include <hip/hip_bf16.h>
#include <math.h>

// Problem constants (from reference)
#define BB 4096
#define KK 50
#define EE 128
#define KPAD 64          // 50 neighbors padded to 4 WMMA row tiles
#define STR 264          // LDS row stride in halfs (256 + 8 pad, keeps 16B alignment)
#define WSLOT (128 * STR)            // halfs per staged weight image
#define WS_BYTES (5u * WSLOT * 2u)   // total f16 weight workspace bytes

typedef __attribute__((ext_vector_type(16))) _Float16 v16h;
typedef __attribute__((ext_vector_type(8)))  _Float16 v8h;
typedef __attribute__((ext_vector_type(4)))  _Float16 v4h;
typedef __attribute__((ext_vector_type(8)))  float    v8f;

// ---------------------------------------------------------------------------
// CDNA5 async global->LDS copy (ASYNCcnt path, ISA 08_async_tensor §4).
// 16 B per lane per instruction = 512 B per wave-op, no VGPR data staging.
// ---------------------------------------------------------------------------
__device__ __forceinline__ void async_ld_b128(unsigned lds_off,
                                              unsigned long long gaddr) {
    asm volatile("global_load_async_to_lds_b128 %0, %1, off"
                 :: "v"(lds_off), "v"(gaddr)
                 : "memory");
}
__device__ __forceinline__ void wait_async0() {
    asm volatile("s_wait_asynccnt 0x0" ::: "memory");
}

// ---- WMMA A/B fragment loader (16-bit A 16x32 layout, ISA 7.12.2) ----
// lane L<16 : row = tile+L,    K = ks*32 + {0..7, 16..23}
// lane L>=16: row = tile+L-16, K = ks*32 + {8..15, 24..31}
__device__ __forceinline__ v16h ldfrag(const _Float16* __restrict__ buf,
                                       int tile16, int ks, int lane) {
    const int r  = tile16 + (lane & 15);
    const int kb = ks * 32 + ((lane >> 4) << 3);
    const _Float16* p = buf + r * STR + kb;
    v8h lo = *(const v8h*)(p);        // ds_load_b128
    v8h hi = *(const v8h*)(p + 16);   // ds_load_b128
    v16h f;
#pragma unroll
    for (int i = 0; i < 8; ++i) { f[i] = lo[i]; f[i + 8] = hi[i]; }
    return f;
}

// ---- one fused Linear+ReLU layer on the matrix pipes ----
template <int KSTEPS>
__device__ __forceinline__ void gemm_layer(const _Float16* __restrict__ Abuf,
                                           const _Float16* __restrict__ Wt,
                                           const float* __restrict__ bias,
                                           _Float16* __restrict__ Obuf,
                                           int ocol, int wave, int lane) {
    const int rtile = (wave >> 1) << 4;   // 0,16,32,48
    const int cbase = (wave & 1) * 4;     // col tiles [cbase, cbase+4)
    v8f acc[4] = {};
#pragma unroll
    for (int ks = 0; ks < KSTEPS; ++ks) {
        v16h a = ldfrag(Abuf, rtile, ks, lane);
#pragma unroll
        for (int c = 0; c < 4; ++c) {
            v16h w = ldfrag(Wt, (cbase + c) << 4, ks, lane);
            acc[c] = __builtin_amdgcn_wmma_f32_16x16x32_f16(
                false, a, false, w, (short)0, acc[c], false, false);
        }
    }
    const int rofs = rtile + ((lane >> 4) << 3);
#pragma unroll
    for (int c = 0; c < 4; ++c) {
        const int col = ((cbase + c) << 4) + (lane & 15);
        const float bv = bias[col];
#pragma unroll
        for (int r = 0; r < 8; ++r) {
            float v = acc[c][r] + bv;
            v = v > 0.f ? v : 0.f;                       // ReLU
            Obuf[(rofs + r) * STR + ocol + col] = (_Float16)v;
        }
    }
}

// Stage one pre-converted f16 weight image (global, padded layout) into LDS.
__device__ __forceinline__ void stage_async(_Float16* __restrict__ Wt,
                                            const _Float16* __restrict__ wq,
                                            int tid) {
    const unsigned lds_base = (unsigned)(size_t)(void*)Wt;
    const unsigned long long gbase = (unsigned long long)(size_t)wq;
    // 67584 B = 4224 x 16 B chunks
    for (unsigned off = (unsigned)tid * 16u; off < WSLOT * 2u; off += 256u * 16u)
        async_ld_b128(lds_base + off, gbase + off);
    wait_async0();
}

// Fallback: stage W[k][128] f32 (row-major) -> Wt[n][k] f16 (transposed)
__device__ __forceinline__ void stageW_f32(_Float16* __restrict__ Wt,
                                           const float* __restrict__ w,
                                           int Kdim, int tid) {
    for (int i = tid; i < 128 * Kdim; i += 256) {
        const int n = i & 127;
        const int k = i >> 7;
        Wt[n * STR + k] = (_Float16)w[k * 128 + n];
    }
}

// ---------------------------------------------------------------------------
// Prep kernel: convert the 5 weight matrices (f32 row-major [K,128]) into
// transposed, LDS-padded f16 images in d_ws, once per launch.
// ---------------------------------------------------------------------------
__global__ void __launch_bounds__(256)
convert_weights_kernel(const float* __restrict__ w_fus,
                       const float* __restrict__ w_c1,
                       const float* __restrict__ w_c2,
                       const float* __restrict__ w_a1,
                       const float* __restrict__ w_a2,
                       _Float16* __restrict__ ws) {
    const int m = blockIdx.y;
    const float* src; _Float16* dst; int Kdim;
    switch (m) {
        case 0:  src = w_fus; dst = ws + 0 * WSLOT; Kdim = 256; break;
        case 1:  src = w_c1;  dst = ws + 1 * WSLOT; Kdim = 256; break;
        case 2:  src = w_c2;  dst = ws + 2 * WSLOT; Kdim = 128; break;
        case 3:  src = w_a1;  dst = ws + 3 * WSLOT; Kdim = 256; break;
        default: src = w_a2;  dst = ws + 4 * WSLOT; Kdim = 128; break;
    }
    const int total = 128 * Kdim;
    for (int i = blockIdx.x * 256 + threadIdx.x; i < total; i += gridDim.x * 256) {
        const int n = i & 127;
        const int k = i >> 7;
        dst[n * STR + k] = (_Float16)src[k * 128 + n];   // coalesced global read
    }
}

// ---------------------------------------------------------------------------
// Main kernel: one block per batch element b; 8 wave32s; 5 WMMA layers.
// ---------------------------------------------------------------------------
__global__ void __launch_bounds__(256)
influence_prop_kernel(const float* __restrict__ u_embs,
                      const float* __restrict__ i_embs,
                      const int*   __restrict__ act_users,
                      const float* __restrict__ user_emb_table,
                      const float* __restrict__ user_profiles,
                      const float* __restrict__ w_fus, const float* __restrict__ b_fus,
                      const float* __restrict__ w_c1,  const float* __restrict__ b_c1,
                      const float* __restrict__ w_c2,  const float* __restrict__ b_c2,
                      const float* __restrict__ w_a1,  const float* __restrict__ b_a1,
                      const float* __restrict__ w_a2,  const float* __restrict__ b_a2,
                      const float* __restrict__ w_a3,  const float* __restrict__ b_a3,
                      const _Float16* __restrict__ wq,   // pre-converted weights or null
                      float* __restrict__ out_combined,
                      float* __restrict__ out_att) {
    extern __shared__ char smem[];
    _Float16* A0  = (_Float16*)smem;            // 64 x STR
    _Float16* A1  = A0 + KPAD * STR;            // 64 x STR
    _Float16* Wt  = A1 + KPAD * STR;            // 128 x STR
    float*    sc  = (float*)(Wt + 128 * STR);   // 64 scores
    float*    att = sc + KPAD;                  // 64 attention weights

    const int b    = blockIdx.x;
    const int tid  = threadIdx.x;
    const int wave = tid >> 5;
    const int lane = tid & 31;
    const bool use_wq = (wq != nullptr);

    // warm L2 with the (heavily reused) weight images
    if (use_wq && tid < 160)
        __builtin_prefetch(wq + (size_t)tid * (5 * WSLOT / 160), 0, 1);

    // ---- gather: A0 rows = [neigh_emb || neigh_prof], f32->f16, float4 loads
    for (int i = tid; i < KPAD * 64; i += 256) {
        const int row = i >> 6;
        const int c4  = (i & 63) << 2;
        float4 v = {0.f, 0.f, 0.f, 0.f};
        if (row < KK) {
            const int idx = act_users[b * KK + row];
            const float* sp = (c4 < 128)
                ? (user_emb_table + (size_t)idx * 128 + c4)
                : (user_profiles  + (size_t)idx * 128 + (c4 - 128));
            v = *(const float4*)sp;               // global_load_b128
        }
        v4h h; h[0] = (_Float16)v.x; h[1] = (_Float16)v.y;
               h[2] = (_Float16)v.z; h[3] = (_Float16)v.w;
        *(v4h*)(A0 + row * STR + c4) = h;         // ds_store_b64
    }
    // A1 cols 128..255 = i_emb broadcast
    for (int i = tid; i < KPAD * 32; i += 256) {
        const int row = i >> 5;
        const int c4  = (i & 31) << 2;
        float4 v = *(const float4*)(i_embs + (size_t)b * 128 + c4);
        v4h h; h[0] = (_Float16)v.x; h[1] = (_Float16)v.y;
               h[2] = (_Float16)v.z; h[3] = (_Float16)v.w;
        *(v4h*)(A1 + row * STR + 128 + c4) = h;
    }

    // ---- L1: fused = relu([emb||prof] @ w_fus + b) -> A1 cols 0..127
    if (use_wq) stage_async(Wt, wq + 0 * WSLOT, tid);
    else        stageW_f32(Wt, w_fus, 256, tid);
    __syncthreads();
    gemm_layer<8>(A0, Wt, b_fus, A1, 0, wave, lane);
    __syncthreads();

    // ---- L2: coup1 = relu([fused||i_emb] @ w_c1 + b) -> A0 cols 0..127
    if (use_wq) stage_async(Wt, wq + 1 * WSLOT, tid);
    else        stageW_f32(Wt, w_c1, 256, tid);
    __syncthreads();
    gemm_layer<8>(A1, Wt, b_c1, A0, 0, wave, lane);
    __syncthreads();

    // ---- L3: coup2 = relu(coup1 @ w_c2 + b) -> A1 cols 0..127
    // simultaneously repurpose A1 cols 128..255 for u_emb broadcast
    for (int i = tid; i < KPAD * 32; i += 256) {
        const int row = i >> 5;
        const int c4  = (i & 31) << 2;
        float4 v = *(const float4*)(u_embs + (size_t)b * 128 + c4);
        v4h h; h[0] = (_Float16)v.x; h[1] = (_Float16)v.y;
               h[2] = (_Float16)v.z; h[3] = (_Float16)v.w;
        *(v4h*)(A1 + row * STR + 128 + c4) = h;
    }
    if (use_wq) stage_async(Wt, wq + 2 * WSLOT, tid);
    else        stageW_f32(Wt, w_c2, 128, tid);
    __syncthreads();
    gemm_layer<4>(A0, Wt, b_c2, A1, 0, wave, lane);
    __syncthreads();

    // ---- L4: h1 = relu([coup2||u_emb] @ w_a1 + b) -> A0 cols 0..127
    if (use_wq) stage_async(Wt, wq + 3 * WSLOT, tid);
    else        stageW_f32(Wt, w_a1, 256, tid);
    __syncthreads();
    gemm_layer<8>(A1, Wt, b_a1, A0, 0, wave, lane);
    __syncthreads();

    // ---- L5: h2 = relu(h1 @ w_a2 + b) -> A0 cols 128..255 (disjoint cols)
    if (use_wq) stage_async(Wt, wq + 4 * WSLOT, tid);
    else        stageW_f32(Wt, w_a2, 128, tid);
    __syncthreads();
    gemm_layer<4>(A0, Wt, b_a2, A0, 128, wave, lane);
    __syncthreads();

    // ---- scores: s[k] = h2[k] . w_a3 + b_a3
    if (tid < KPAD) {
        float s = b_a3[0];
#pragma unroll 4
        for (int e = 0; e < 128; ++e)
            s += (float)A0[tid * STR + 128 + e] * w_a3[e];
        sc[tid] = (tid < KK) ? s : -1e30f;
    }
    __syncthreads();

    // ---- softmax over K=50 neighbors (wave 0, wave32 shuffles)
    if (tid < 32) {
        float v0 = sc[tid];
        float v1 = sc[tid + 32];
        float m = fmaxf(v0, v1);
#pragma unroll
        for (int off = 16; off > 0; off >>= 1)
            m = fmaxf(m, __shfl_xor(m, off, 32));
        float e0 = __expf(v0 - m);
        float e1 = __expf(v1 - m);
        float s = e0 + e1;
#pragma unroll
        for (int off = 16; off > 0; off >>= 1)
            s += __shfl_xor(s, off, 32);
        const float inv = 1.f / s;
        att[tid]      = e0 * inv;
        att[tid + 32] = e1 * inv;
    }
    __syncthreads();

    if (tid < KK) out_att[b * KK + tid] = att[tid];

    // ---- combined[e] = sum_k coup2[k][e] * att[k]  (coup2 in A1 cols 0..127)
    if (tid < 128) {
        float acc = 0.f;
#pragma unroll 2
        for (int k = 0; k < KK; ++k)
            acc += (float)A1[k * STR + tid] * att[k];
        out_combined[b * 128 + tid] = acc;
    }
}

extern "C" void kernel_launch(void* const* d_in, const int* in_sizes, int n_in,
                              void* d_out, int out_size, void* d_ws, size_t ws_size,
                              hipStream_t stream) {
    // setup_inputs order:
    // 0 users, 1 u_embs, 2 items, 3 i_embs, 4 act_users, 5 user_emb_table,
    // 6 user_profiles, 7 w_fus, 8 b_fus, 9 w_c1, 10 b_c1, 11 w_c2, 12 b_c2,
    // 13 w_a1, 14 b_a1, 15 w_a2, 16 b_a2, 17 w_a3, 18 b_a3
    const float* u_embs         = (const float*)d_in[1];
    const float* i_embs         = (const float*)d_in[3];
    const int*   act_users      = (const int*)  d_in[4];
    const float* user_emb_table = (const float*)d_in[5];
    const float* user_profiles  = (const float*)d_in[6];
    const float* w_fus = (const float*)d_in[7];  const float* b_fus = (const float*)d_in[8];
    const float* w_c1  = (const float*)d_in[9];  const float* b_c1  = (const float*)d_in[10];
    const float* w_c2  = (const float*)d_in[11]; const float* b_c2  = (const float*)d_in[12];
    const float* w_a1  = (const float*)d_in[13]; const float* b_a1  = (const float*)d_in[14];
    const float* w_a2  = (const float*)d_in[15]; const float* b_a2  = (const float*)d_in[16];
    const float* w_a3  = (const float*)d_in[17]; const float* b_a3  = (const float*)d_in[18];

    float* out_combined = (float*)d_out;            // [B, E]
    float* out_att      = out_combined + BB * EE;   // [B, K, 1]

    _Float16* wq = nullptr;
    if (ws_size >= (size_t)WS_BYTES) {
        wq = (_Float16*)d_ws;
        convert_weights_kernel<<<dim3(16, 5), 256, 0, stream>>>(
            w_fus, w_c1, w_c2, w_a1, w_a2, wq);
    }

    const size_t smem_bytes =
        (size_t)(2 * KPAD * STR + 128 * STR) * sizeof(_Float16) + 2 * KPAD * sizeof(float);

    hipFuncSetAttribute((const void*)influence_prop_kernel,
                        hipFuncAttributeMaxDynamicSharedMemorySize,
                        (int)smem_bytes);

    influence_prop_kernel<<<BB, 256, smem_bytes, stream>>>(
        u_embs, i_embs, act_users, user_emb_table, user_profiles,
        w_fus, b_fus, w_c1, b_c1, w_c2, b_c2,
        w_a1, b_a1, w_a2, b_a2, w_a3, b_a3,
        wq, out_combined, out_att);
}
